// NPairsLoss_48713519072039
// MI455X (gfx1250) — matrix operand order repaired and checked by reading, not compiled, not run on hardware
//
#include <hip/hip_runtime.h>
#include <hip/hip_bf16.h>

typedef __attribute__((ext_vector_type(16))) _Float16 v16h;
typedef __attribute__((ext_vector_type(8)))  _Float16 v8h;
typedef __attribute__((ext_vector_type(8)))  float    v8f;

#define B_     1024
#define P_     4
#define D_     512
#define NROW   4096     // B_*P_
#define NTILE  256      // NROW/16 column tiles
#define KSTEPS 16       // D_/32 WMMA K-steps
#define MSTRIP 32       // rows per workgroup strip (2 A tiles)
#define NBLK   (NROW / MSTRIP)   // 128 workgroups

// ---------------- kernel 0: zero output, convert fp32 -> f16 ----------------
__global__ void npairs_convert_kernel(const float* __restrict__ X,
                                      _Float16* __restrict__ Xh,
                                      float* __restrict__ out) {
    int i = blockIdx.x * blockDim.x + threadIdx.x;
    if (i == 0) out[0] = 0.0f;
    const int total = NROW * D_;
    for (int idx = i; idx < total; idx += gridDim.x * blockDim.x)
        Xh[idx] = (_Float16)X[idx];
}

// 32x16 f16 B fragment: two 16-byte chunks per lane at +0 and +16 elems (2x global b128).
__device__ __forceinline__ v16h load_bfrag(const _Float16* __restrict__ p) {
    const v8h lo = *(const v8h*)p;
    const v8h hi = *(const v8h*)(p + 16);
    v16h r;
#pragma unroll
    for (int e = 0; e < 8; ++e) { r[e] = lo[e]; r[e + 8] = hi[e]; }
    return r;
}

// 16x32 f16 A fragment from LDS (2x ds b128). tile t = strip rows 16t..16t+15.
__device__ __forceinline__ v16h load_afrag(const _Float16* __restrict__ a_s,
                                           int l15, int half, int t, int kt) {
    const _Float16* p = a_s + (l15 + 16 * t) * D_ + kt * 32 + 8 * half;
    const v8h lo = *(const v8h*)p;
    const v8h hi = *(const v8h*)(p + 16 - 8 * half + 8 * half); // +16 stripe
    v16h r;
    const v8h hi2 = *(const v8h*)(p + 16);
#pragma unroll
    for (int e = 0; e < 8; ++e) { r[e] = lo[e]; r[e + 8] = hi2[e]; }
    (void)hi;
    return r;
}

// ---------------- kernel 1: fused strip GEMM + masked loss ----------------
// One workgroup per 32-row strip; 32x4096 prod strip stashed in LDS as f16
// (256KB of CDNA5's 320KB). 2x2 register tiling: each B fragment feeds 2
// WMMAs and vice versa; 4 independent accumulator chains hide load latency
// at our LDS-bound occupancy of 1 wave/SIMD.
__global__ void __launch_bounds__(128, 1)
npairs_loss_kernel(const _Float16* __restrict__ Xh,
                   const int* __restrict__ tgt,
                   float* __restrict__ out) {
    extern __shared__ char smem_raw[];
    _Float16* prod_s = (_Float16*)smem_raw;                           // 32*4096 f16 = 256KB
    _Float16* a_s    = (_Float16*)(smem_raw + MSTRIP * NROW * 2);     // 32*512 f16 = 32KB
    int*      tgt_s  = (int*)((char*)a_s + MSTRIP * D_ * 2);          // 1024 i32 = 4KB
    float*    S_s    = (float*)((char*)tgt_s + B_ * 4);               // 32 f32
    float*    loss_s = S_s + MSTRIP;                                  // 1 f32

    const int tid  = threadIdx.x;
    const int lane = tid & 31;
    const int wave = tid >> 5;
    const int half = lane >> 4;
    const int l15  = lane & 15;
    const int m0   = blockIdx.x * MSTRIP;

    if (tid < MSTRIP) S_s[tid] = 0.0f;
    if (tid == 0) loss_s[0] = 0.0f;

    {   // A strip (32 rows x 512 f16) -> LDS
        const v8h* src = (const v8h*)(Xh + (size_t)m0 * D_);
        v8h*       dst = (v8h*)a_s;
        for (int i = tid; i < (MSTRIP * D_) / 8; i += 128) dst[i] = src[i];
    }
    for (int i = tid; i < B_; i += 128) tgt_s[i] = tgt[i];

    __syncthreads();

    float sacc[2][8];
#pragma unroll
    for (int t = 0; t < 2; ++t)
#pragma unroll
        for (int j = 0; j < 8; ++j) sacc[t][j] = 0.0f;

    const int ntBeg = wave * (NTILE / 4);
    const int ntEnd = ntBeg + (NTILE / 4);

    // ---- pass 1: (A0,A1) x (B0,B1) per step, single-stage prefetch ----
    for (int nt = ntBeg; nt < ntEnd; nt += 2) {
        const int col0 = nt * 16 + l15;
        const int col1 = col0 + 16;
        const _Float16* brow0 = Xh + (size_t)col0 * D_ + half * 8;
        const _Float16* brow1 = Xh + (size_t)col1 * D_ + half * 8;
        if (nt + 2 < ntEnd) {
            __builtin_prefetch(Xh + (size_t)(col0 + 32) * D_, 0, 0);
            __builtin_prefetch(Xh + (size_t)(col1 + 32) * D_, 0, 0);
        }

        v8f acc00 = {}, acc01 = {}, acc10 = {}, acc11 = {};
        v16h a0 = load_afrag(a_s, l15, half, 0, 0);
        v16h a1 = load_afrag(a_s, l15, half, 1, 0);
        v16h b0 = load_bfrag(brow0);
        v16h b1 = load_bfrag(brow1);
#pragma unroll
        for (int kt = 0; kt < KSTEPS; ++kt) {
            v16h na0 = a0, na1 = a1, nb0 = b0, nb1 = b1;
            if (kt + 1 < KSTEPS) {           // issue next-K loads before WMMAs
                na0 = load_afrag(a_s, l15, half, 0, kt + 1);
                na1 = load_afrag(a_s, l15, half, 1, kt + 1);
                nb0 = load_bfrag(brow0 + (kt + 1) * 32);
                nb1 = load_bfrag(brow1 + (kt + 1) * 32);
            }
            acc00 = __builtin_amdgcn_wmma_f32_16x16x32_f16(false, a0, false, b0, (short)0, acc00, false, false);
            acc10 = __builtin_amdgcn_wmma_f32_16x16x32_f16(false, a1, false, b0, (short)0, acc10, false, false);
            acc01 = __builtin_amdgcn_wmma_f32_16x16x32_f16(false, a0, false, b1, (short)0, acc01, false, false);
            acc11 = __builtin_amdgcn_wmma_f32_16x16x32_f16(false, a1, false, b1, (short)0, acc11, false, false);
            a0 = na0; a1 = na1; b0 = nb0; b1 = nb1;
        }

        // C layout: VGPR j -> M = 16t + j + 8*half, N = lane&15. Stash + S.
        const int tc0 = tgt_s[col0 >> 2], pc0 = col0 & 3;
        const int tc1 = tgt_s[col1 >> 2], pc1 = col1 & 3;
#pragma unroll
        for (int t = 0; t < 2; ++t) {
            const v8f& ac0 = t ? acc10 : acc00;
            const v8f& ac1 = t ? acc11 : acc01;
#pragma unroll
            for (int j = 0; j < 8; ++j) {
                const int M  = 16 * t + j + 8 * half;
                const int ig = m0 + M;
                const int ti = tgt_s[ig >> 2];
                const int pi = ig & 3;
                prod_s[M * NROW + col0] = (_Float16)ac0[j];
                prod_s[M * NROW + col1] = (_Float16)ac1[j];
                if (ti != tc0 && pi != pc0) sacc[t][j] += __expf(ac0[j]);  // dadc
                if (ti != tc1 && pi != pc1) sacc[t][j] += __expf(ac1[j]);
            }
        }
    }

    // S reduction: 16-lane halves via xor shuffles, then cross-wave LDS atomics.
#pragma unroll
    for (int t = 0; t < 2; ++t)
#pragma unroll
        for (int j = 0; j < 8; ++j) {
            float v = sacc[t][j];
#pragma unroll
            for (int m = 1; m < 16; m <<= 1) v += __shfl_xor(v, m, 32);
            if (l15 == 0) atomicAdd(&S_s[16 * t + j + 8 * half], v);
        }

    __syncthreads();   // prod strip + S complete

    // ---- pass 2: masked log(1 + S_i * exp(-p)) reduction over the strip ----
    float lacc = 0.0f;
    for (int e = tid; e < MSTRIP * NROW; e += 128) {
        const int row = e >> 12;          // / 4096
        const int col = e & (NROW - 1);
        const int ig  = m0 + row;
        const bool sc = (tgt_s[ig >> 2] == tgt_s[col >> 2]);
        const bool sp = ((ig & 3) == (col & 3));
        float w = 0.0f;
        if (!sc &&  sp) w = 2.0f;         // s_sadc counted twice (reference bug kept)
        if ( sc && !sp) w = 1.0f;         // s_dasc
        if (w != 0.0f) {
            const float p = (float)prod_s[row * NROW + col];
            lacc += w * __logf(fmaf(S_s[row], __expf(-p), 1.0f));
        }
    }
#pragma unroll
    for (int m = 1; m < 32; m <<= 1) lacc += __shfl_xor(lacc, m, 32);
    if (lane == 0) atomicAdd(loss_s, lacc);
    __syncthreads();
    if (tid == 0) atomicAdd(out, loss_s[0] * (1.0f / (float)NROW));
}

extern "C" void kernel_launch(void* const* d_in, const int* in_sizes, int n_in,
                              void* d_out, int out_size, void* d_ws, size_t ws_size,
                              hipStream_t stream) {
    const float* X   = (const float*)d_in[0];   // (1024, 4, 512) fp32
    const int*   tgt = (const int*)d_in[1];     // (1024,) labels
    float*       out = (float*)d_out;           // scalar loss
    _Float16*    Xh  = (_Float16*)d_ws;         // 4096*512 f16 = 4MB scratch

    npairs_convert_kernel<<<1024, 256, 0, stream>>>(X, Xh, out);

    const size_t smem = (size_t)MSTRIP * NROW * 2   // prod strip (f16) 256KB
                      + (size_t)MSTRIP * D_ * 2     // A strip          32KB
                      + (size_t)B_ * 4              // targets           4KB
                      + MSTRIP * 4 + 4;             // S + block loss
    npairs_loss_kernel<<<NBLK, 128, smem, stream>>>(Xh, tgt, out);
}
